// AdapterFusion_82008105549863
// MI455X (gfx1250) — matrix operand order repaired and checked by reading, not compiled, not run on hardware
//
#include <hip/hip_runtime.h>

// ---------------------------------------------------------------------------
// AdapterFusion fused pipeline for MI455X (gfx1250, wave32, WMMA).
//
// Math: probs = softmax(query·(Wq^T Wk)·key^T / 50), over N=8
//       ctx   = (sum_n probs_n * value_n) @ Wv^T + residual
// The k-projection (155 GFLOP) is folded into M = Wq^T Wk (0.9 GFLOP once),
// making the kernel memory-bound on the 805 MB key/value stream (~39 us at
// 23.3 TB/s HBM). GEMMs use v_wmma_f32_16x16x32_bf16 with f32 accumulation;
// all GEMMs are C = A·B^T so LDS staging is pure b128 (via
// GLOBAL_LOAD_ASYNC_TO_LDS_B128 when the builtin is available).
// ---------------------------------------------------------------------------

typedef __attribute__((ext_vector_type(16))) __bf16 v16bf;
typedef __attribute__((ext_vector_type(8)))  float  v8f;
typedef __attribute__((ext_vector_type(4)))  float  f32x4;
typedef __attribute__((ext_vector_type(4)))  int    v4i;

static constexpr int Dm = 768;       // feature dim
static constexpr int Bn = 8;         // batch
static constexpr int Sn = 2048;      // sequence
static constexpr int Nn = 8;         // adapters
static constexpr int BS = Bn * Sn;   // 16384 token rows
static constexpr float INV_T = 1.0f / 50.0f;

#if defined(__has_builtin)
#if __has_builtin(__builtin_amdgcn_global_load_async_to_lds_b128) && \
    __has_builtin(__builtin_amdgcn_s_wait_asynccnt)
#define USE_ASYNC_LDS 1
#endif
#endif

// 16B global -> LDS stage (async DMA path when available)
__device__ __forceinline__ void stage16(__bf16* lds_dst, const __bf16* g_src) {
#ifdef USE_ASYNC_LDS
  __builtin_amdgcn_global_load_async_to_lds_b128(
      (__attribute__((address_space(1))) v4i*)(v4i*)const_cast<__bf16*>(g_src),
      (__attribute__((address_space(3))) v4i*)(v4i*)lds_dst,
      0, 0);
#else
  *reinterpret_cast<uint4*>(lds_dst) = *reinterpret_cast<const uint4*>(g_src);
#endif
}

__device__ __forceinline__ void stage_fence() {
#ifdef USE_ASYNC_LDS
  __builtin_amdgcn_s_wait_asynccnt(0);
#endif
}

// ---------------- conversion kernels ----------------

__global__ void cvt_f32_to_bf16_x4(const float* __restrict__ in,
                                   __bf16* __restrict__ out) {
  size_t i = ((size_t)blockIdx.x * blockDim.x + threadIdx.x) * 4;
  float4 f = *reinterpret_cast<const float4*>(in + i);
  out[i + 0] = (__bf16)f.x;
  out[i + 1] = (__bf16)f.y;
  out[i + 2] = (__bf16)f.z;
  out[i + 3] = (__bf16)f.w;
}

// out[d*dim + i] = (bf16) in[i*dim + d]   (32x32 LDS tiles, no bank conflicts)
__global__ void transpose_cvt_bf16(const float* __restrict__ in,
                                   __bf16* __restrict__ out, int dim) {
  __shared__ float tile[32][33];
  const int bx = blockIdx.x * 32;
  const int by = blockIdx.y * 32;
  const int c  = threadIdx.x & 31;
  const int r0 = threadIdx.x >> 5;
#pragma unroll
  for (int p = 0; p < 4; ++p) {
    int r = r0 + p * 8;
    tile[r][c] = in[(size_t)(by + r) * dim + bx + c];
  }
  __syncthreads();
#pragma unroll
  for (int p = 0; p < 4; ++p) {
    int r = r0 + p * 8;
    out[(size_t)(bx + r) * dim + by + c] = (__bf16)tile[c][r];
  }
}

// ---------------- WMMA GEMM (C = A[M,K] * B[N,K]^T) ----------------
// 256 threads = 8 waves; block tile 64x64; K-step 32. Each wave computes two
// 16x16 C tiles sharing one A fragment (2 wmma / 6 ds_load_b128 per step).
// Shapes are multiples of the tiles (768, 16384).

__device__ __forceinline__ v16bf load_frag_a(const __bf16* rowp, int h) {
  // 16-bit A 16x32 layout: elems 0..7 -> K=h*8+0..7, elems 8..15 -> K=16+h*8+0..7
  union { v16bf v; uint4 u[2]; } t;
  t.u[0] = *reinterpret_cast<const uint4*>(rowp + h * 8);
  t.u[1] = *reinterpret_cast<const uint4*>(rowp + 16 + h * 8);
  return t.v;
}

__device__ __forceinline__ v16bf load_frag_b(const __bf16* rowp, int h) {
  // 16-bit B 32x16 layout: lane holds column n, elems 0..7 -> K=h*8+0..7, elems 8..15 -> K=16+h*8+0..7
  union { v16bf v; uint4 u[2]; } t;
  t.u[0] = *reinterpret_cast<const uint4*>(rowp + h * 8);
  t.u[1] = *reinterpret_cast<const uint4*>(rowp + 16 + h * 8);
  return t.v;
}

template <bool OUT_BF16, bool ADD_RES>
__global__ void gemm_wmma_bt(const __bf16* __restrict__ A,
                             const __bf16* __restrict__ B,
                             float* __restrict__ Cf,
                             __bf16* __restrict__ Cb,
                             const float* __restrict__ Res,
                             int M, int N, int K) {
  constexpr int TM = 64, TN = 64, TK = 32, LDT = 40;  // 40-elem pad: 16B aligned
  __shared__ __align__(16) __bf16 As[TM * LDT];
  __shared__ __align__(16) __bf16 Bs[TN * LDT];

  const int m0   = blockIdx.x * TM;
  const int n0   = blockIdx.y * TN;
  const int tid  = threadIdx.x;
  const int wave = tid >> 5;
  const int lane = tid & 31;
  const int wm   = (wave & 3) * 16;   // 4 waves along M
  const int wn   = (wave >> 2) * 32;  // 2 waves along N, 2 tiles each

  v8f acc0 = {};
  v8f acc1 = {};

  const int sr = tid >> 2, sc = (tid & 3) << 3;  // staging coords (64 rows x 4 b128)
  for (int k0 = 0; k0 < K; k0 += TK) {
    __syncthreads();
    stage16(&As[sr * LDT + sc], A + (size_t)(m0 + sr) * K + k0 + sc);
    stage16(&Bs[sr * LDT + sc], B + (size_t)(n0 + sr) * K + k0 + sc);
    stage_fence();
    __syncthreads();

    const int r = lane & 15, h = lane >> 4;
    v16bf af = load_frag_a(&As[(wm + r) * LDT], h);
    v16bf b0 = load_frag_b(&Bs[(wn + r) * LDT], h);
    v16bf b1 = load_frag_b(&Bs[(wn + 16 + r) * LDT], h);
    acc0 = __builtin_amdgcn_wmma_f32_16x16x32_bf16(false, af, false, b0,
                                                   (short)0, acc0, false, false);
    acc1 = __builtin_amdgcn_wmma_f32_16x16x32_bf16(false, af, false, b1,
                                                   (short)0, acc1, false, false);
  }

  // C/D layout: VGPR v -> row (wm + (lane>=16 ? 8 : 0) + v), col = lane%16
  const int nn0   = n0 + wn + (lane & 15);
  const int mbase = m0 + wm + ((lane >> 4) << 3);
#pragma unroll
  for (int v = 0; v < 8; ++v) {
    size_t i0 = (size_t)(mbase + v) * N + nn0;
    size_t i1 = i0 + 16;
    float v0 = acc0[v], v1 = acc1[v];
    if constexpr (ADD_RES) { v0 += Res[i0]; v1 += Res[i1]; }
    if constexpr (OUT_BF16) { Cb[i0] = (__bf16)v0; Cb[i1] = (__bf16)v1; }
    else                    { Cf[i0] = v0;         Cf[i1] = v1; }
  }
}

// ---------------- fused scores/softmax/value-mix ----------------
// One wave32 per (b,s) token. Streams key+value (805 MB) exactly once with
// non-temporal b128 loads so GEMM weights stay resident in the 192 MB L2.

__device__ __forceinline__ float wave_reduce_bcast(float v) {
#pragma unroll
  for (int off = 16; off > 0; off >>= 1) v += __shfl_down(v, off, 32);
  return __shfl(v, 0, 32);
}

struct __align__(8) bf16x4 { __bf16 v[4]; };

__global__ void attn_softmax_mix(const __bf16* __restrict__ U,
                                 const float* __restrict__ key,
                                 const float* __restrict__ value,
                                 float* __restrict__ probs_out,
                                 __bf16* __restrict__ ctx_raw) {
  const int pair = blockIdx.x * 8 + (threadIdx.x >> 5);  // token row (b*S+s)
  const int lane = threadIdx.x & 31;
  const int col  = lane * 4;                             // 4 consecutive elems/lane

  const __bf16* u    = U + (size_t)pair * Dm;
  const float* kbase = key   + (size_t)pair * Nn * Dm;
  const float* vbase = value + (size_t)pair * Nn * Dm;

  float ur[6][4];
#pragma unroll
  for (int j = 0; j < 6; ++j) {
    const __bf16* up = u + j * 128 + col;
#pragma unroll
    for (int c = 0; c < 4; ++c) ur[j][c] = (float)up[c];
  }

  float score[Nn];
#pragma unroll
  for (int n = 0; n < Nn; ++n) {
    const float* kp = kbase + n * Dm;
    float acc = 0.f;
#pragma unroll
    for (int j = 0; j < 6; ++j) {
      f32x4 k4 = __builtin_nontemporal_load(
          reinterpret_cast<const f32x4*>(kp + j * 128 + col));
      acc = fmaf(ur[j][0], k4[0], acc);
      acc = fmaf(ur[j][1], k4[1], acc);
      acc = fmaf(ur[j][2], k4[2], acc);
      acc = fmaf(ur[j][3], k4[3], acc);
    }
    score[n] = wave_reduce_bcast(acc);
  }

  // softmax(scores / T) — redundant in all lanes (sums already broadcast)
  float mx = score[0];
#pragma unroll
  for (int n = 1; n < Nn; ++n) mx = fmaxf(mx, score[n]);
  float p[Nn], s = 0.f;
#pragma unroll
  for (int n = 0; n < Nn; ++n) { p[n] = __expf((score[n] - mx) * INV_T); s += p[n]; }
  const float inv = 1.0f / s;
#pragma unroll
  for (int n = 0; n < Nn; ++n) p[n] *= inv;

  if (lane == 0) {
#pragma unroll
    for (int n = 0; n < Nn; ++n) probs_out[(size_t)pair * Nn + n] = p[n];
  }

  // ctx_raw[pair] = sum_n p[n] * value[pair, n, :]
#pragma unroll
  for (int j = 0; j < 6; ++j) {
    f32x4 a4 = {0.f, 0.f, 0.f, 0.f};
#pragma unroll
    for (int n = 0; n < Nn; ++n) {
      f32x4 vv = __builtin_nontemporal_load(
          reinterpret_cast<const f32x4*>(vbase + n * Dm + j * 128 + col));
      a4 += vv * p[n];
    }
    bf16x4 o;
#pragma unroll
    for (int c = 0; c < 4; ++c) o.v[c] = (__bf16)a4[c];
    *reinterpret_cast<bf16x4*>(ctx_raw + (size_t)pair * Dm + j * 128 + col) = o;
  }
}

// ---------------- host launch ----------------

extern "C" void kernel_launch(void* const* d_in, const int* in_sizes, int n_in,
                              void* d_out, int out_size, void* d_ws, size_t ws_size,
                              hipStream_t stream) {
  (void)in_sizes; (void)n_in; (void)out_size; (void)ws_size;
  const float* query    = (const float*)d_in[0];  // [B,S,D]
  const float* key      = (const float*)d_in[1];  // [B,S,N,D]
  const float* value    = (const float*)d_in[2];  // [B,S,N,D]
  const float* residual = (const float*)d_in[3];  // [B,S,D]
  const float* Wq       = (const float*)d_in[4];  // [D,D]
  const float* Wk       = (const float*)d_in[6];  // [D,D]  (bq=d_in[5], bk=d_in[7] zero)
  const float* Wv       = (const float*)d_in[8];  // [D,D]

  float* probs = (float*)d_out;                    // [B,S,N]
  float* ctx   = (float*)d_out + (size_t)BS * Nn;  // [B,S,D]

  // workspace (bf16); ctx_raw aliases q_b (q dead after the u GEMM)
  char* ws = (char*)d_ws;
  __bf16* q_b   = (__bf16*)(ws);                           // 25,165,824 B
  __bf16* WqT_b = (__bf16*)(ws + 25165824);                //  1,179,648 B
  __bf16* WkT_b = (__bf16*)(ws + 25165824 + 1179648);
  __bf16* Wv_b  = (__bf16*)(ws + 25165824 + 2 * 1179648);
  __bf16* MT_b  = (__bf16*)(ws + 25165824 + 3 * 1179648);
  __bf16* u_b   = (__bf16*)(ws + 25165824 + 4 * 1179648);  // 25,165,824 B
  __bf16* ctx_raw = q_b;                                   // alias

  // 1) precision conversions / transposes
  cvt_f32_to_bf16_x4<<<(BS * Dm) / 1024, 256, 0, stream>>>(query, q_b);
  cvt_f32_to_bf16_x4<<<(Dm * Dm) / 1024, 256, 0, stream>>>(Wv, Wv_b);
  transpose_cvt_bf16<<<dim3(Dm / 32, Dm / 32), 256, 0, stream>>>(Wq, WqT_b, Dm);
  transpose_cvt_bf16<<<dim3(Dm / 32, Dm / 32), 256, 0, stream>>>(Wk, WkT_b, Dm);

  // 2) M^T[e,d] = sum_i Wk[i,e]*Wq[i,d]  ==  WkT @ WqT^T   (768^3, bf16 out)
  gemm_wmma_bt<true, false><<<dim3(Dm / 64, Dm / 64), 256, 0, stream>>>(
      WkT_b, WqT_b, nullptr, MT_b, nullptr, Dm, Dm, Dm);

  // 3) u[m,e] = sum_d q[m,d]*M[d,e]  ==  q @ MT^T   (16384x768x768, bf16 out)
  gemm_wmma_bt<true, false><<<dim3(BS / 64, Dm / 64), 256, 0, stream>>>(
      q_b, MT_b, nullptr, u_b, nullptr, BS, Dm, Dm);

  // 4) scores + softmax + value mix (streams key/value once, NT b128 loads)
  attn_softmax_mix<<<BS / 8, 256, 0, stream>>>(u_b, key, value, probs, ctx_raw);

  // 5) ctx = ctx_raw @ Wv^T + residual  (f32 out straight to d_out)
  gemm_wmma_bt<false, true><<<dim3(BS / 64, Dm / 64), 256, 0, stream>>>(
      ctx_raw, Wv_b, ctx, nullptr, residual, BS, Dm, Dm);
}
